// Net_78443282695009
// MI455X (gfx1250) — compile-verified, matrix operand order checked by reading, not run on hardware
//
#include <hip/hip_runtime.h>
#include <hip/hip_bf16.h>

// ---------------------------------------------------------------------------
// PointNet++-style network on MI455X (gfx1250).
// All GEMMs run as v_wmma_f32_16x16x32_f16 (f16 in, f32 accumulate).
// Compute-bound workload (~33 GFLOP over a few MB -> L2 resident), so the
// whole pipeline is organized around WMMA tiles; segment-max exploits the
// fact that e*_dst is a tiled arange (64 consecutive edges per destination).
// ---------------------------------------------------------------------------

typedef __attribute__((ext_vector_type(8)))  _Float16 v8h;
typedef __attribute__((ext_vector_type(16))) _Float16 v16h;
typedef __attribute__((ext_vector_type(8)))  float    v8f;

#define WMMA_F16(a, b, c) \
    __builtin_amdgcn_wmma_f32_16x16x32_f16(false, (a), false, (b), (short)0, (c), false, false)

// Load one 16x32 f16 WMMA fragment (A layout; B^T uses the identical layout).
// Per ISA 7.12.2: lane L holds row M = L%16; lanes<16 cover K in
// {k..k+7, k+16..k+23}, lanes>=16 cover {k+8..k+15, k+24..k+31}.
// With row-major storage that is two contiguous 16-byte chunks per lane.
__device__ __forceinline__ v16h load_frag(const _Float16* base, int row, int ld, int k) {
    int lane = threadIdx.x & 31;
    int r  = row + (lane & 15);
    int lh = lane >> 4;
    const _Float16* p = base + (size_t)r * ld + k + 8 * lh;
    v8h lo = *(const v8h*)(p);
    v8h hi = *(const v8h*)(p + 16);
    v16h out;
#pragma unroll
    for (int i = 0; i < 8; ++i) { out[i] = lo[i]; out[i + 8] = hi[i]; }
    return out;
}

// ---------------------------------------------------------------------------
// Weight prep: out[n][k] = (k < fi) ? w[k][n] : 0   (f32 (fi x fo) -> f16 (fo x kpad))
// Transposed + K-padded so B fragments load exactly like A fragments.
// ---------------------------------------------------------------------------
__global__ void convert_wT_kernel(const float* __restrict__ w, _Float16* __restrict__ wT,
                                  int fi, int fo, int kpad) {
    int idx = blockIdx.x * blockDim.x + threadIdx.x;
    if (idx >= fo * kpad) return;
    int n = idx / kpad;
    int k = idx - n * kpad;
    wT[idx] = (k < fi) ? (_Float16)w[(size_t)k * fo + n] : (_Float16)0.f;
}

__global__ void gather_pos1_kernel(const float* __restrict__ pos, const int* __restrict__ idx1,
                                   float* __restrict__ pos1, int n3) {
    int i = blockIdx.x * blockDim.x + threadIdx.x;
    if (i >= n3) return;
    int r = i / 3, c = i - 3 * r;
    pos1[i] = pos[(size_t)idx1[r] * 3 + c];
}

// ---------------------------------------------------------------------------
// Stage 1: per destination q (8192 blocks), 64 edges, MLP 3->64->64->128,
// mask -> -inf, max over the 64 edge rows, relu, store x1 (f16).
// Block = 128 threads = 4 waves; wave w owns rows 16w..16w+15.
// ---------------------------------------------------------------------------
__global__ void stage1_kernel(const float* __restrict__ pos, const int* __restrict__ idx1,
                              const int* __restrict__ e1_src, const int* __restrict__ e1_mask,
                              const _Float16* __restrict__ w1aT, const float* __restrict__ b1a,
                              const _Float16* __restrict__ w1bT, const float* __restrict__ b1b,
                              const _Float16* __restrict__ w1cT, const float* __restrict__ b1c,
                              _Float16* __restrict__ x1f16) {
    __shared__ __align__(16) _Float16 Af[64][32];
    __shared__ __align__(16) _Float16 H1[64][64];
    __shared__ __align__(16) _Float16 H2[64][64];
    __shared__ float red[4][128];
    __shared__ int   smask[64];

    const int q   = blockIdx.x;
    const int tid = threadIdx.x;
    const int p1  = idx1[q];
    const float qx = pos[3 * p1 + 0], qy = pos[3 * p1 + 1], qz = pos[3 * p1 + 2];

    if (tid < 64) {
        int e = q * 64 + tid;
        int s = e1_src[e];
        smask[tid] = e1_mask[e];
        Af[tid][0] = (_Float16)(pos[3 * s + 0] - qx);
        Af[tid][1] = (_Float16)(pos[3 * s + 1] - qy);
        Af[tid][2] = (_Float16)(pos[3 * s + 2] - qz);
#pragma unroll
        for (int k = 3; k < 32; ++k) Af[tid][k] = (_Float16)0.f;
    }
    __syncthreads();

    const int wave = tid >> 5, lane = tid & 31;
    const int row0 = wave * 16;
    const int lh = lane >> 4, cl = lane & 15;

    // ---- L1: 64x32 @ 32x64 ----
    {
        v16h a = load_frag(&Af[0][0], row0, 32, 0);
        v8f acc[4] = {};
#pragma unroll
        for (int nt = 0; nt < 4; ++nt) {
            v16h b = load_frag(w1aT, nt * 16, 32, 0);
            acc[nt] = WMMA_F16(a, b, acc[nt]);
        }
#pragma unroll
        for (int nt = 0; nt < 4; ++nt) {
            int col = nt * 16 + cl;
            float bv = b1a[col];
#pragma unroll
            for (int r = 0; r < 8; ++r)
                H1[row0 + r + 8 * lh][col] = (_Float16)fmaxf(acc[nt][r] + bv, 0.f);
        }
    }
    __syncthreads();

    // ---- L2: 64x64 @ 64x64 ----
    {
        v8f acc[4] = {};
        for (int k = 0; k < 64; k += 32) {
            v16h a = load_frag(&H1[0][0], row0, 64, k);
#pragma unroll
            for (int nt = 0; nt < 4; ++nt) {
                v16h b = load_frag(w1bT, nt * 16, 64, k);
                acc[nt] = WMMA_F16(a, b, acc[nt]);
            }
        }
#pragma unroll
        for (int nt = 0; nt < 4; ++nt) {
            int col = nt * 16 + cl;
            float bv = b1b[col];
#pragma unroll
            for (int r = 0; r < 8; ++r)
                H2[row0 + r + 8 * lh][col] = (_Float16)fmaxf(acc[nt][r] + bv, 0.f);
        }
    }
    __syncthreads();

    // ---- L3: 64x64 @ 64x128, then masked max over 64 rows ----
    {
        v8f acc[8] = {};
        for (int k = 0; k < 64; k += 32) {
            v16h a = load_frag(&H2[0][0], row0, 64, k);
#pragma unroll
            for (int nt = 0; nt < 8; ++nt) {
                v16h b = load_frag(w1cT, nt * 16, 64, k);
                acc[nt] = WMMA_F16(a, b, acc[nt]);
            }
        }
#pragma unroll
        for (int nt = 0; nt < 8; ++nt) {
            int col = nt * 16 + cl;
            float bv = b1c[col];
            float m = -3.0e38f;
#pragma unroll
            for (int r = 0; r < 8; ++r) {
                int row = row0 + r + 8 * lh;
                float v = acc[nt][r] + bv;
                if (smask[row]) m = fmaxf(m, v);
            }
            m = fmaxf(m, __shfl_xor(m, 16, 32));
            if (lh == 0) red[wave][col] = m;
        }
    }
    __syncthreads();
    if (tid < 128) {
        float v = fmaxf(fmaxf(red[0][tid], red[1][tid]), fmaxf(red[2][tid], red[3][tid]));
        x1f16[(size_t)q * 128 + tid] = (_Float16)fmaxf(v, 0.f);  // relu(segment_max)
    }
}

// ---------------------------------------------------------------------------
// Stage 2: per destination q (2048 blocks), 64 edges, feat = [x1[src] | rel],
// MLP 131(->160 pad)->128->128->256, mask+max+relu, write directly into the
// padded global-MLP feature buffer gfeat[q] = [x2(256) | pos2(3) | 0..(288)].
// ---------------------------------------------------------------------------
__global__ void stage2_kernel(const float* __restrict__ pos1, const int* __restrict__ idx2,
                              const int* __restrict__ e2_src, const int* __restrict__ e2_mask,
                              const _Float16* __restrict__ x1f16,
                              const _Float16* __restrict__ w2aT, const float* __restrict__ b2a,
                              const _Float16* __restrict__ w2bT, const float* __restrict__ b2b,
                              const _Float16* __restrict__ w2cT, const float* __restrict__ b2c,
                              _Float16* __restrict__ gfeat) {
    __shared__ __align__(16) _Float16 A2[64][160];
    __shared__ __align__(16) _Float16 H1[64][128];
    __shared__ __align__(16) _Float16 H2[64][128];
    __shared__ float red[4][64];
    __shared__ int   smask[64];
    __shared__ int   ssrc[64];
    __shared__ float srel[64][3];

    const int q   = blockIdx.x;
    const int tid = threadIdx.x;
    const int i2  = idx2[q];
    const float qx = pos1[3 * i2 + 0], qy = pos1[3 * i2 + 1], qz = pos1[3 * i2 + 2];

    if (tid < 64) {
        int e = q * 64 + tid;
        int s = e2_src[e];
        ssrc[tid]  = s;
        smask[tid] = e2_mask[e];
        srel[tid][0] = pos1[3 * s + 0] - qx;
        srel[tid][1] = pos1[3 * s + 1] - qy;
        srel[tid][2] = pos1[3 * s + 2] - qz;
    }
    __syncthreads();

    for (int idx = tid; idx < 64 * 160; idx += 128) {
        int row = idx / 160, col = idx - row * 160;
        _Float16 v;
        if (col < 128)      v = x1f16[(size_t)ssrc[row] * 128 + col];
        else if (col < 131) v = (_Float16)srel[row][col - 128];
        else                v = (_Float16)0.f;
        A2[row][col] = v;
    }
    __syncthreads();

    const int wave = tid >> 5, lane = tid & 31;
    const int row0 = wave * 16;
    const int lh = lane >> 4, cl = lane & 15;

    // ---- L1: 64x160 @ 160x128 ----
    {
        v8f acc[8] = {};
        for (int k = 0; k < 160; k += 32) {
            v16h a = load_frag(&A2[0][0], row0, 160, k);
#pragma unroll
            for (int nt = 0; nt < 8; ++nt) {
                v16h b = load_frag(w2aT, nt * 16, 160, k);
                acc[nt] = WMMA_F16(a, b, acc[nt]);
            }
        }
#pragma unroll
        for (int nt = 0; nt < 8; ++nt) {
            int col = nt * 16 + cl;
            float bv = b2a[col];
#pragma unroll
            for (int r = 0; r < 8; ++r)
                H1[row0 + r + 8 * lh][col] = (_Float16)fmaxf(acc[nt][r] + bv, 0.f);
        }
    }
    __syncthreads();

    // ---- L2: 64x128 @ 128x128 ----
    {
        v8f acc[8] = {};
        for (int k = 0; k < 128; k += 32) {
            v16h a = load_frag(&H1[0][0], row0, 128, k);
#pragma unroll
            for (int nt = 0; nt < 8; ++nt) {
                v16h b = load_frag(w2bT, nt * 16, 128, k);
                acc[nt] = WMMA_F16(a, b, acc[nt]);
            }
        }
#pragma unroll
        for (int nt = 0; nt < 8; ++nt) {
            int col = nt * 16 + cl;
            float bv = b2b[col];
#pragma unroll
            for (int r = 0; r < 8; ++r)
                H2[row0 + r + 8 * lh][col] = (_Float16)fmaxf(acc[nt][r] + bv, 0.f);
        }
    }
    __syncthreads();

    // ---- L3: 64x128 @ 128x256 in four 64-col chunks, masked max + relu ----
    for (int nc = 0; nc < 4; ++nc) {
        v8f acc[4] = {};
        for (int k = 0; k < 128; k += 32) {
            v16h a = load_frag(&H2[0][0], row0, 128, k);
#pragma unroll
            for (int nt = 0; nt < 4; ++nt) {
                v16h b = load_frag(w2cT, nc * 64 + nt * 16, 128, k);
                acc[nt] = WMMA_F16(a, b, acc[nt]);
            }
        }
#pragma unroll
        for (int nt = 0; nt < 4; ++nt) {
            int coll = nt * 16 + cl;
            float bv = b2c[nc * 64 + coll];
            float m = -3.0e38f;
#pragma unroll
            for (int r = 0; r < 8; ++r) {
                int row = row0 + r + 8 * lh;
                float v = acc[nt][r] + bv;
                if (smask[row]) m = fmaxf(m, v);
            }
            m = fmaxf(m, __shfl_xor(m, 16, 32));
            if (lh == 0) red[wave][coll] = m;
        }
        __syncthreads();
        if (tid < 64) {
            float v = fmaxf(fmaxf(red[0][tid], red[1][tid]), fmaxf(red[2][tid], red[3][tid]));
            gfeat[(size_t)q * 288 + nc * 64 + tid] = (_Float16)fmaxf(v, 0.f);
        }
        __syncthreads();
    }
    // pos2 + zero padding -> cols 256..287
    if (tid < 32) {
        _Float16 v = (_Float16)0.f;
        if (tid == 0) v = (_Float16)qx;
        else if (tid == 1) v = (_Float16)qy;
        else if (tid == 2) v = (_Float16)qz;
        gfeat[(size_t)q * 288 + 256 + tid] = v;
    }
}

// ---------------------------------------------------------------------------
// Generic 64x64-tile WMMA GEMM: C = act(A @ B + bias)
// A: f16 (M x lda) row-major; BT: f16 (N x ldb) = B transposed, ldb == Kpad.
// ---------------------------------------------------------------------------
__global__ void gemm64_kernel(const _Float16* __restrict__ A, int lda,
                              const _Float16* __restrict__ BT, int ldb,
                              const float* __restrict__ bias,
                              _Float16* __restrict__ Cf16, float* __restrict__ Cf32,
                              int ldc, int K, int relu) {
    const int wave = threadIdx.x >> 5, lane = threadIdx.x & 31;
    const int row0 = blockIdx.x * 64 + wave * 16;
    const int col0 = blockIdx.y * 64;
    const int lh = lane >> 4, cl = lane & 15;

    v8f acc[4] = {};
    for (int k = 0; k < K; k += 32) {
        if (k + 32 < K) __builtin_prefetch(A + (size_t)row0 * lda + k + 32, 0, 1);
        v16h a = load_frag(A, row0, lda, k);
#pragma unroll
        for (int nt = 0; nt < 4; ++nt) {
            v16h b = load_frag(BT, col0 + nt * 16, ldb, k);
            acc[nt] = WMMA_F16(a, b, acc[nt]);
        }
    }
#pragma unroll
    for (int nt = 0; nt < 4; ++nt) {
        int col = col0 + nt * 16 + cl;
        float bv = bias ? bias[col] : 0.f;
#pragma unroll
        for (int r = 0; r < 8; ++r) {
            int row = row0 + r + 8 * lh;
            float v = acc[nt][r] + bv;
            if (relu) v = fmaxf(v, 0.f);
            if (Cf16) Cf16[(size_t)row * ldc + col] = (_Float16)v;
            if (Cf32) Cf32[(size_t)row * ldc + col] = v;
        }
    }
}

// Max over 128 consecutive rows: g3out (2048 x 1024) -> gmax (16 x 1024)
__global__ void reduce_max128_kernel(const float* __restrict__ g3out, float* __restrict__ gmax) {
    int i = blockIdx.x * blockDim.x + threadIdx.x;
    if (i >= 16 * 1024) return;
    int b = i >> 10, col = i & 1023;
    const float* p = g3out + (size_t)b * 128 * 1024 + col;
    float m = -3.0e38f;
    for (int r = 0; r < 128; ++r) m = fmaxf(m, p[(size_t)r * 1024]);
    gmax[i] = m;
}

// Head: 16x1024 -> 512 -> 256 -> 10 + log_softmax (tiny; plain f32)
__global__ void head_kernel(const float* __restrict__ gmax,
                            const float* __restrict__ l1w, const float* __restrict__ l1b,
                            const float* __restrict__ l2w, const float* __restrict__ l2b,
                            const float* __restrict__ l3w, const float* __restrict__ l3b,
                            float* __restrict__ out) {
    __shared__ float h1[16][512];
    __shared__ float h2[16][256];
    __shared__ float logit[16][10];
    const int tid = threadIdx.x;

    for (int idx = tid; idx < 16 * 512; idx += 256) {
        int row = idx >> 9, col = idx & 511;
        float s = l1b[col];
        const float* g = gmax + (size_t)row * 1024;
        for (int k = 0; k < 1024; ++k) s += g[k] * l1w[(size_t)k * 512 + col];
        h1[row][col] = fmaxf(s, 0.f);
    }
    __syncthreads();
    for (int idx = tid; idx < 16 * 256; idx += 256) {
        int row = idx >> 8, col = idx & 255;
        float s = l2b[col];
        for (int k = 0; k < 512; ++k) s += h1[row][k] * l2w[(size_t)k * 256 + col];
        h2[row][col] = fmaxf(s, 0.f);
    }
    __syncthreads();
    for (int idx = tid; idx < 160; idx += 256) {
        int row = idx / 10, col = idx - row * 10;
        float s = l3b[col];
        for (int k = 0; k < 256; ++k) s += h2[row][k] * l3w[(size_t)k * 10 + col];
        logit[row][col] = s;
    }
    __syncthreads();
    if (tid < 16) {
        float m = -3.0e38f;
        for (int c = 0; c < 10; ++c) m = fmaxf(m, logit[tid][c]);
        float sum = 0.f;
        for (int c = 0; c < 10; ++c) sum += expf(logit[tid][c] - m);
        float lse = m + logf(sum);
        for (int c = 0; c < 10; ++c) out[tid * 10 + c] = logit[tid][c] - lse;
    }
}

// ---------------------------------------------------------------------------
// Launcher
// ---------------------------------------------------------------------------
extern "C" void kernel_launch(void* const* d_in, const int* in_sizes, int n_in,
                              void* d_out, int out_size, void* d_ws, size_t ws_size,
                              hipStream_t stream) {
    (void)in_sizes; (void)n_in; (void)out_size; (void)ws_size;

    const float* pos     = (const float*)d_in[0];
    const int*   idx1    = (const int*)d_in[1];
    const int*   e1_src  = (const int*)d_in[2];
    // d_in[3] = e1_dst unused: dst == edge/64 structurally
    const int*   e1_mask = (const int*)d_in[4];
    const int*   idx2    = (const int*)d_in[5];
    const int*   e2_src  = (const int*)d_in[6];
    // d_in[7] = e2_dst unused
    const int*   e2_mask = (const int*)d_in[8];
    const float* w1a = (const float*)d_in[9];  const float* b1a = (const float*)d_in[10];
    const float* w1b = (const float*)d_in[11]; const float* b1b = (const float*)d_in[12];
    const float* w1c = (const float*)d_in[13]; const float* b1c = (const float*)d_in[14];
    const float* w2a = (const float*)d_in[15]; const float* b2a = (const float*)d_in[16];
    const float* w2b = (const float*)d_in[17]; const float* b2b = (const float*)d_in[18];
    const float* w2c = (const float*)d_in[19]; const float* b2c = (const float*)d_in[20];
    const float* g1w = (const float*)d_in[21]; const float* g1b = (const float*)d_in[22];
    const float* g2w = (const float*)d_in[23]; const float* g2b = (const float*)d_in[24];
    const float* g3w = (const float*)d_in[25]; const float* g3b = (const float*)d_in[26];
    const float* l1w = (const float*)d_in[27]; const float* l1b = (const float*)d_in[28];
    const float* l2w = (const float*)d_in[29]; const float* l2b = (const float*)d_in[30];
    const float* l3w = (const float*)d_in[31]; const float* l3b = (const float*)d_in[32];
    float* out = (float*)d_out;

    // Workspace carve-up (~16.6 MB total)
    char* ws = (char*)d_ws;
    size_t off = 0;
    auto alloc = [&](size_t bytes) -> void* {
        void* p = ws + off;
        off = (off + bytes + 255) & ~(size_t)255;
        return p;
    };
    _Float16* w1aT  = (_Float16*)alloc(64   * 32  * 2);
    _Float16* w1bT  = (_Float16*)alloc(64   * 64  * 2);
    _Float16* w1cT  = (_Float16*)alloc(128  * 64  * 2);
    _Float16* w2aT  = (_Float16*)alloc(128  * 160 * 2);
    _Float16* w2bT  = (_Float16*)alloc(128  * 128 * 2);
    _Float16* w2cT  = (_Float16*)alloc(256  * 128 * 2);
    _Float16* g1wT  = (_Float16*)alloc(256  * 288 * 2);
    _Float16* g2wT  = (_Float16*)alloc(512  * 256 * 2);
    _Float16* g3wT  = (_Float16*)alloc(1024 * 512 * 2);
    float*    pos1  = (float*)   alloc(8192 * 3   * 4);
    _Float16* x1f16 = (_Float16*)alloc((size_t)8192 * 128 * 2);
    _Float16* gfeat = (_Float16*)alloc((size_t)2048 * 288 * 2);
    _Float16* hg1   = (_Float16*)alloc((size_t)2048 * 256 * 2);
    _Float16* hg2   = (_Float16*)alloc((size_t)2048 * 512 * 2);
    float*    g3out = (float*)   alloc((size_t)2048 * 1024 * 4);
    float*    gmax  = (float*)   alloc(16 * 1024 * 4);

    // Weight conversions: f32 (fi x fo) -> f16 transposed (fo x kpad)
    convert_wT_kernel<<<(64   * 32  + 255) / 256, 256, 0, stream>>>(w1a, w1aT, 3,   64,   32);
    convert_wT_kernel<<<(64   * 64  + 255) / 256, 256, 0, stream>>>(w1b, w1bT, 64,  64,   64);
    convert_wT_kernel<<<(128  * 64  + 255) / 256, 256, 0, stream>>>(w1c, w1cT, 64,  128,  64);
    convert_wT_kernel<<<(128  * 160 + 255) / 256, 256, 0, stream>>>(w2a, w2aT, 131, 128,  160);
    convert_wT_kernel<<<(128  * 128 + 255) / 256, 256, 0, stream>>>(w2b, w2bT, 128, 128,  128);
    convert_wT_kernel<<<(256  * 128 + 255) / 256, 256, 0, stream>>>(w2c, w2cT, 128, 256,  128);
    convert_wT_kernel<<<(256  * 288 + 255) / 256, 256, 0, stream>>>(g1w, g1wT, 259, 256,  288);
    convert_wT_kernel<<<(512  * 256 + 255) / 256, 256, 0, stream>>>(g2w, g2wT, 256, 512,  256);
    convert_wT_kernel<<<(1024 * 512 + 255) / 256, 256, 0, stream>>>(g3w, g3wT, 512, 1024, 512);

    gather_pos1_kernel<<<(8192 * 3 + 255) / 256, 256, 0, stream>>>(pos, idx1, pos1, 8192 * 3);

    stage1_kernel<<<8192, 128, 0, stream>>>(pos, idx1, e1_src, e1_mask,
                                            w1aT, b1a, w1bT, b1b, w1cT, b1c, x1f16);

    stage2_kernel<<<2048, 128, 0, stream>>>(pos1, idx2, e2_src, e2_mask, x1f16,
                                            w2aT, b2a, w2bT, b2b, w2cT, b2c, gfeat);

    // Global MLP: 2048 x (288 -> 256 -> 512 -> 1024)
    gemm64_kernel<<<dim3(2048 / 64, 256 / 64),  128, 0, stream>>>(gfeat, 288, g1wT, 288, g1b,
                                                                  hg1, nullptr, 256, 288, 1);
    gemm64_kernel<<<dim3(2048 / 64, 512 / 64),  128, 0, stream>>>(hg1, 256, g2wT, 256, g2b,
                                                                  hg2, nullptr, 512, 256, 1);
    gemm64_kernel<<<dim3(2048 / 64, 1024 / 64), 128, 0, stream>>>(hg2, 512, g3wT, 512, g3b,
                                                                  nullptr, g3out, 1024, 512, 0);

    reduce_max128_kernel<<<(16 * 1024 + 255) / 256, 256, 0, stream>>>(g3out, gmax);

    head_kernel<<<1, 256, 0, stream>>>(gmax, l1w, l1b, l2w, l2b, l3w, l3b, out);
}
